// TimeSeriesEmbedding_20057497273050
// MI455X (gfx1250) — compile-verified, matrix operand order checked by reading, not run on hardware
//
#include <hip/hip_runtime.h>
#include <hip/hip_bf16.h>
#include <math.h>

typedef __bf16 bf16_t;
typedef __attribute__((ext_vector_type(16))) __bf16 v16bf;
typedef __attribute__((ext_vector_type(8)))  float  v8f;

typedef unsigned int u32;
typedef __attribute__((ext_vector_type(4))) u32 tdm_v4u;
typedef __attribute__((ext_vector_type(8))) int tdm_v8i;
typedef __attribute__((ext_vector_type(4))) int tdm_v4i;

#define PATCH   16
#define EMB     16
#define HIDN    1024
#define BATCH   128
#define SEQ     4096
#define PMAXN   256          // SEQ / PATCH
#define IN_DIM  272          // PATCH + PATCH*EMB
#define KPAD    288          // IN_DIM padded to multiple of 32
#define MROWS   (BATCH * PMAXN)  // 32768

#if __has_builtin(__builtin_amdgcn_tensor_load_to_lds) && \
    __has_builtin(__builtin_amdgcn_s_wait_tensorcnt)
#define USE_TDM 1
#else
#define USE_TDM 0
#endif

// ---------------------------------------------------------------------------
// Stage 1: featurization -> bf16 feat[MROWS][KPAD]
// ---------------------------------------------------------------------------
__global__ void featurize_kernel(const float* __restrict__ x,
                                 const int*   __restrict__ lengths,
                                 const float* __restrict__ pos_table,
                                 bf16_t*      __restrict__ feat) {
    int r = blockIdx.x;            // 0..MROWS-1
    int c = threadIdx.x;           // 0..KPAD-1  (288 = 9 waves)
    int b = r >> 8;
    int p = r & 255;
    int len = lengths[b];
    float v;
    if (c < PATCH) {
        int t  = p * PATCH + c;
        int gi = min(t, max(len - 1, 0));
        v = x[((size_t)b * SEQ + gi) * 2 + 0];
    } else if (c < IN_DIM) {
        int jj = c - PATCH;
        int j  = jj >> 4;
        int e  = jj & 15;
        int t  = p * PATCH + j;
        int pos = (t < len) ? t : SEQ;      // PAD_IDX = SEQ
        v = pos_table[pos * EMB + e];
    } else {
        v = 0.0f;                            // K padding
    }
    feat[(size_t)r * KPAD + c] = (bf16_t)v;
}

// ---------------------------------------------------------------------------
// Weight transpose + f32 -> bf16 conversion (N-major, K contiguous)
// ---------------------------------------------------------------------------
__global__ void w1_transpose_kernel(const float* __restrict__ W1,
                                    bf16_t* __restrict__ w1t) {
    int n = blockIdx.x;            // 0..HIDN-1
    int k = threadIdx.x;           // 0..KPAD-1
    float v = (k < IN_DIM) ? W1[(size_t)k * HIDN + n] : 0.0f;
    w1t[(size_t)n * KPAD + k] = (bf16_t)v;
}

__global__ void w2_transpose_kernel(const float* __restrict__ W2,
                                    bf16_t* __restrict__ w2t) {
    int n = blockIdx.x;            // 0..HIDN-1
    for (int k = threadIdx.x; k < HIDN; k += 256) {
        w2t[(size_t)n * HIDN + k] = (bf16_t)W2[(size_t)k * HIDN + n];
    }
}

// ---------------------------------------------------------------------------
// n_patch output (int32 bit pattern written into tail of d_out)
// ---------------------------------------------------------------------------
__global__ void npatch_kernel(const int* __restrict__ lengths,
                              int* __restrict__ np) {
    int b = threadIdx.x;
    if (b < BATCH) np[b] = (lengths[b] + PATCH - 1) / PATCH;
}

#if USE_TDM
// ---------------------------------------------------------------------------
// LDS byte-offset of a __shared__ object (addrspace(3) cast).
// ---------------------------------------------------------------------------
__device__ __forceinline__ u32 lds_byte_off(const void* p) {
    return (u32)(unsigned long long)
        (__attribute__((address_space(3))) const char*)p;
}

// ---------------------------------------------------------------------------
// Issue a TDM load of a 2D bf16 tile [rows=128 x cols=32] into LDS.
// D# layout per CDNA5 ISA:
//   group0: count=1 | lds_addr | global_addr[56:0] | type=2
//   group1: data_size=1 (2B), tensor_dim0=stride, tensor_dim1=big,
//           tile_dim0=32, tile_dim1=128, tensor_dim0_stride=stride
//   trailing groups: zero (2D tile, no iterate/pad/gather)
// 6-arg builtin form (clang-23): (v4u, v8i, v4i, v4i, v8i, i32 cpol).
// Completion tracked by TENSORcnt.
// ---------------------------------------------------------------------------
__device__ __forceinline__ void tdm_load_tile_2d(u32 ldsoff,
                                                 const bf16_t* gtile,
                                                 u32 stride_elems) {
    unsigned long long ga = (unsigned long long)(const void*)gtile;
    tdm_v4u g0;
    g0.x = 1u;                                        // count=1, user mode
    g0.y = ldsoff;                                    // lds_addr (bytes)
    g0.z = (u32)ga;                                   // global_addr[31:0]
    g0.w = (u32)((ga >> 32) & 0x01FFFFFFu)            // global_addr[56:32]
         | (2u << 30);                                // type=2 (image)
    tdm_v8i g1;
    g1[0] = (int)(1u << 16);                          // data_size=1 -> 2 bytes
    g1[1] = (int)((stride_elems & 0xFFFFu) << 16);    // tensor_dim0[15:0]
    g1[2] = (int)((stride_elems >> 16) & 0xFFFFu);    // tensor_dim0[31:16] | dim1[15:0]=0
    g1[3] = (int)(0x0010u | (32u << 16));             // tensor_dim1[31:16]=0x10 (1M rows) | tile_dim0=32
    g1[4] = (int)(128u);                              // tile_dim1=128 | tile_dim2=0
    g1[5] = (int)stride_elems;                        // tensor_dim0_stride[31:0]
    g1[6] = 0;                                        // stride[47:32]=0 | dim1_stride[15:0]=0
    g1[7] = 0;                                        // dim1_stride[47:16]=0
    tdm_v4i z4 = {0, 0, 0, 0};
    tdm_v8i z8 = {0, 0, 0, 0, 0, 0, 0, 0};
    __builtin_amdgcn_tensor_load_to_lds(g0, g1, z4, z4, z8, 0);
}
#endif // USE_TDM

// ---------------------------------------------------------------------------
// WMMA fragment loader from a 16x32 bf16 tile (row stride 32) in LDS.
// 16-bit A/B layout (ISA 7.12.2): lane<16 holds row=lane, K 0..7 (v0..3)
// and K 16..23 (v4..7); lane>=16 holds row=lane-16, K 8..15 and K 24..31.
// ---------------------------------------------------------------------------
__device__ __forceinline__ v16bf load_frag(const bf16_t* base, int lane) {
    int row = lane & 15;
    int k0  = (lane >> 4) << 3;   // 0 or 8
    union { v16bf v; uint4 q[2]; } u;
    u.q[0] = *(const uint4*)(base + row * 32 + k0);
    u.q[1] = *(const uint4*)(base + row * 32 + k0 + 16);
    return u.v;
}

// ---------------------------------------------------------------------------
// Tiled bf16 WMMA GEMM:  C[M x N] = A[M x K] * Bt[N x K]^T  (+bias, epilogue)
// Block tile 128x128, BK = 32, 256 threads = 8 waves, wave tile 32x64.
// Double-buffered LDS fed by the Tensor Data Mover (wave0: A, wave1: B),
// synchronized with s_wait_tensorcnt + workgroup barrier.
// ---------------------------------------------------------------------------
template<int KSTEPS, bool DO_GELU>
__global__ void __launch_bounds__(256, 1)
gemm_wmma_kernel(const bf16_t* __restrict__ A,
                 const bf16_t* __restrict__ Bt,
                 const float*  __restrict__ bias,
                 const int*    __restrict__ lengths,
                 bf16_t*       __restrict__ outBf,
                 float*        __restrict__ outF,
                 int lda, int ldb) {
    __shared__ __align__(16) bf16_t sA[2][128 * 32];
    __shared__ __align__(16) bf16_t sB[2][128 * 32];

    const int tid  = threadIdx.x;
    const int lane = tid & 31;
    const int wid  = tid >> 5;      // 0..7
    const int wm   = wid & 3;       // 4 waves along M (32 rows each)
    const int wn   = wid >> 2;      // 2 waves along N (64 cols each)
    const int bm   = blockIdx.y;    // M / 128
    const int bn   = blockIdx.x;    // N / 128

    v8f acc[2][4];
#pragma unroll
    for (int mi = 0; mi < 2; ++mi)
#pragma unroll
        for (int ni = 0; ni < 4; ++ni)
            acc[mi][ni] = (v8f){0.f,0.f,0.f,0.f,0.f,0.f,0.f,0.f};

    const bf16_t* Abase = A  + (size_t)(bm * 128) * lda;
    const bf16_t* Bbase = Bt + (size_t)(bn * 128) * ldb;

#if USE_TDM
    const u32 offA[2] = { lds_byte_off(&sA[0][0]), lds_byte_off(&sA[1][0]) };
    const u32 offB[2] = { lds_byte_off(&sB[0][0]), lds_byte_off(&sB[1][0]) };

    // Prologue: DMA K-step 0 into buffer 0.
    if (wid == 0)      tdm_load_tile_2d(offA[0], Abase, (u32)lda);
    else if (wid == 1) tdm_load_tile_2d(offB[0], Bbase, (u32)ldb);

    for (int ks = 0; ks < KSTEPS; ++ks) {
        const int cur = ks & 1;
        // Issue next K-step's DMA into the other buffer (last read 1 step
        // ago; the trailing barrier of the previous step makes this safe),
        // then wait for the current step's DMA to land.
        if (ks + 1 < KSTEPS) {
            if (wid == 0)
                tdm_load_tile_2d(offA[cur ^ 1], Abase + (ks + 1) * 32, (u32)lda);
            else if (wid == 1)
                tdm_load_tile_2d(offB[cur ^ 1], Bbase + (ks + 1) * 32, (u32)ldb);
            __builtin_amdgcn_s_wait_tensorcnt(1);   // current tile done
        } else {
            __builtin_amdgcn_s_wait_tensorcnt(0);   // last tile done
        }
        __syncthreads();                            // tile visible to all waves

        v16bf af[2], bfr[4];
#pragma unroll
        for (int mi = 0; mi < 2; ++mi)
            af[mi] = load_frag(&sA[cur][(wm * 32 + mi * 16) * 32], lane);
#pragma unroll
        for (int ni = 0; ni < 4; ++ni)
            bfr[ni] = load_frag(&sB[cur][(wn * 64 + ni * 16) * 32], lane);

#pragma unroll
        for (int mi = 0; mi < 2; ++mi)
#pragma unroll
            for (int ni = 0; ni < 4; ++ni)
                acc[mi][ni] = __builtin_amdgcn_wmma_f32_16x16x32_bf16(
                    false, af[mi], false, bfr[ni],
                    (short)0, acc[mi][ni], false, false);

        __syncthreads();            // all reads done; buffer reusable
    }
#else // !USE_TDM: cooperative staging fallback
    const int lrow = tid & 127;
    const int lsel = tid >> 7;
    const bf16_t* grow = lsel ? (Bbase + (size_t)lrow * ldb)
                              : (Abase + (size_t)lrow * lda);
    bf16_t* ldst = (lsel ? &sB[0][0] : &sA[0][0]) + lrow * 32;

    uint4 t0, t1, t2, t3;
    {
        const uint4* g4 = (const uint4*)grow;
        t0 = g4[0]; t1 = g4[1]; t2 = g4[2]; t3 = g4[3];
    }
    for (int ks = 0; ks < KSTEPS; ++ks) {
        __syncthreads();
        {
            uint4* l4 = (uint4*)ldst;
            l4[0] = t0; l4[1] = t1; l4[2] = t2; l4[3] = t3;
        }
        __syncthreads();
        if (ks + 1 < KSTEPS) {
            const uint4* gn = (const uint4*)(grow + (ks + 1) * 32);
            t0 = gn[0]; t1 = gn[1]; t2 = gn[2]; t3 = gn[3];
        }
        v16bf af[2], bfr[4];
#pragma unroll
        for (int mi = 0; mi < 2; ++mi)
            af[mi] = load_frag(&sA[0][(wm * 32 + mi * 16) * 32], lane);
#pragma unroll
        for (int ni = 0; ni < 4; ++ni)
            bfr[ni] = load_frag(&sB[0][(wn * 64 + ni * 16) * 32], lane);
#pragma unroll
        for (int mi = 0; mi < 2; ++mi)
#pragma unroll
            for (int ni = 0; ni < 4; ++ni)
                acc[mi][ni] = __builtin_amdgcn_wmma_f32_16x16x32_bf16(
                    false, af[mi], false, bfr[ni],
                    (short)0, acc[mi][ni], false, false);
    }
#endif

    // Epilogue. C layout (ISA 7.12.2): lanes 0..15 -> N=lane, M=r;
    // lanes 16..31 -> N=lane-16, M=r+8. Fully unrolled (static indexing).
    const int coll  = lane & 15;
    const int rbase = (lane >> 4) * 8;
#pragma unroll
    for (int mi = 0; mi < 2; ++mi) {
#pragma unroll
        for (int ni = 0; ni < 4; ++ni) {
            const int colg = bn * 128 + wn * 64 + ni * 16 + coll;
            const float bv = bias[colg];
#pragma unroll
            for (int r = 0; r < 8; ++r) {
                const int rowg = bm * 128 + wm * 32 + mi * 16 + rbase + r;
                float v = acc[mi][ni][r] + bv;
                if (DO_GELU) {
                    v = 0.5f * v * (1.0f + erff(v * 0.70710678118654752440f));
                    outBf[(size_t)rowg * HIDN + colg] = (bf16_t)v;
                } else {
                    const int b  = rowg >> 8;
                    const int p  = rowg & 255;
                    const int np = (lengths[b] + PATCH - 1) >> 4;
                    outF[(size_t)rowg * HIDN + colg] = (p < np) ? v : 0.0f;
                }
            }
        }
    }
}

// ---------------------------------------------------------------------------
// Launcher
// ---------------------------------------------------------------------------
extern "C" void kernel_launch(void* const* d_in, const int* in_sizes, int n_in,
                              void* d_out, int out_size, void* d_ws, size_t ws_size,
                              hipStream_t stream) {
    const float* x         = (const float*)d_in[0];
    const int*   lengths   = (const int*)  d_in[1];
    const float* pos_table = (const float*)d_in[2];
    const float* W1        = (const float*)d_in[3];
    const float* b1        = (const float*)d_in[4];
    const float* W2        = (const float*)d_in[5];
    const float* b2        = (const float*)d_in[6];

    float* out   = (float*)d_out;
    int* np_out  = (int*)(out + (size_t)MROWS * HIDN);  // tail of d_out

    // workspace layout (bf16)
    char* ws = (char*)d_ws;
    size_t o = 0;
    bf16_t* feat = (bf16_t*)(ws + o); o += (size_t)MROWS * KPAD * sizeof(bf16_t);
    bf16_t* w1t  = (bf16_t*)(ws + o); o += (size_t)HIDN  * KPAD * sizeof(bf16_t);
    bf16_t* w2t  = (bf16_t*)(ws + o); o += (size_t)HIDN  * HIDN * sizeof(bf16_t);
    bf16_t* h    = (bf16_t*)(ws + o); o += (size_t)MROWS * HIDN * sizeof(bf16_t);
    (void)ws_size; (void)in_sizes; (void)n_in; (void)out_size;

    // Stage 1: build features (bf16, K padded to 288)
    featurize_kernel<<<MROWS, KPAD, 0, stream>>>(x, lengths, pos_table, feat);

    // Stage 2: weight transpose/convert
    w1_transpose_kernel<<<HIDN, KPAD, 0, stream>>>(W1, w1t);
    w2_transpose_kernel<<<HIDN, 256, 0, stream>>>(W2, w2t);

    // Stage 3: h = gelu(feat @ W1 + b1)   [32768 x 288] x [288 x 1024]
    gemm_wmma_kernel<KPAD / 32, true>
        <<<dim3(HIDN / 128, MROWS / 128), 256, 0, stream>>>(
            feat, w1t, b1, nullptr, h, nullptr, KPAD, KPAD);

    // Stage 4: out = (h @ W2 + b2) * patch_mask   [32768 x 1024] x [1024 x 1024]
    gemm_wmma_kernel<HIDN / 32, false>
        <<<dim3(HIDN / 128, MROWS / 128), 256, 0, stream>>>(
            h, w2t, b2, lengths, nullptr, out, HIDN, HIDN);

    // Stage 5: n_patch
    npatch_kernel<<<1, BATCH, 0, stream>>>(lengths, np_out);
}
